// BaseWindowAttention_54039278519076
// MI455X (gfx1250) — matrix operand order repaired and checked by reading, not compiled, not run on hardware
//
#include <hip/hip_runtime.h>
#include <hip/hip_bf16.h>

// ---------------------------------------------------------------------------
// Fused Swin-style window attention for MI455X (gfx1250, wave32, WMMA).
//   kernel 1: transpose+convert w_qkv / w_out to bf16 [n][k] in d_ws
//   kernel 2: one workgroup per 8x8 window (2048 blocks x 256 threads):
//             QKV GEMM -> windowed attention (bias+softmax) -> out projection
// All matrix math via v_wmma_f32_16x16x32_bf16, fp32 accumulation.
// Streaming data (x in, out) uses non-temporal cache hints so the bf16
// weights stay resident in the 192MB L2 across all 2048 blocks.
// ---------------------------------------------------------------------------

typedef __bf16 bf16;
typedef __attribute__((ext_vector_type(16))) __bf16 v16bf;
typedef __attribute__((ext_vector_type(8)))  __bf16 v8bf;
typedef __attribute__((ext_vector_type(8)))  float  v8f;
typedef __attribute__((ext_vector_type(4)))  float  v4f;

#define HEADS 8
#define DH    32
#define DIM   256
#define NQKV  768
#define WSQ   64      // tokens per window (8x8)

// LDS row strides (bf16 elements), padded to avoid 64-bank conflicts
#define XSTR 264      // x / attn-out rows: 256 + 8
#define QSTR 36       // q,k rows: 32 + 4
#define VSTR 68       // v^T rows: 64 + 4
#define PSTR 68       // softmax-prob strip rows: 64 + 4

// LDS carve (bytes)
#define OFF_XB   0
#define SZ_XB    (WSQ * XSTR * 2)            // 33792 (reused as attn-out)
#define OFF_Q    (OFF_XB + SZ_XB)
#define SZ_QK    (HEADS * WSQ * QSTR * 2)    // 36864
#define OFF_K    (OFF_Q + SZ_QK)
#define OFF_VT   (OFF_K + SZ_QK)
#define SZ_VT    (HEADS * DH * VSTR * 2)     // 34816
#define OFF_PB   (OFF_VT + SZ_VT)
#define SZ_PB    (HEADS * 16 * PSTR * 2)     // 17408
#define OFF_POS  (OFF_PB + SZ_PB)
#define SZ_POS   (15 * 15 * 4)
#define SMEM_BYTES (((OFF_POS + SZ_POS) + 255) & ~255)

// --- fragment helpers -------------------------------------------------------
// A-fragment: two contiguous 8x bf16 (16B) chunks 16 elements apart in K.
// B-fragment: same two 16B chunks, 16 elements apart in K.
__device__ __forceinline__ v16bf ld_frag2(const bf16* p0, const bf16* p1) {
  v8bf lo = *(const v8bf*)p0;
  v8bf hi = *(const v8bf*)p1;
  v16bf r;
#pragma unroll
  for (int i = 0; i < 8; ++i) { r[i] = lo[i]; r[i + 8] = hi[i]; }
  return r;
}

__device__ __forceinline__ v8f wmma_bf16(v16bf a, v16bf b, v8f c) {
  return __builtin_amdgcn_wmma_f32_16x16x32_bf16(
      /*neg_a=*/false, a, /*neg_b=*/false, b,
      /*c_mod=*/(short)0, c, /*reuse_a=*/false, /*reuse_b=*/false);
}

// ---------------------------------------------------------------------------
// Kernel 1: weights -> bf16, transposed to [n][k] so WMMA B-fragments are
// k-contiguous (two 16B loads per lane).  1024 blocks x 256 threads.
// ---------------------------------------------------------------------------
__global__ void prep_weights(const float* __restrict__ w_qkv,
                             const float* __restrict__ w_out,
                             bf16* __restrict__ wqkvT,
                             bf16* __restrict__ woutT) {
  const int n = blockIdx.x;
  const int k = threadIdx.x;                  // 0..255
  if (n < NQKV) {
    wqkvT[(size_t)n * DIM + k] = (bf16)w_qkv[(size_t)k * NQKV + n];
  } else {
    const int nn = n - NQKV;
    woutT[(size_t)nn * DIM + k] = (bf16)w_out[(size_t)k * DIM + nn];
  }
}

// ---------------------------------------------------------------------------
// Kernel 2: fused window attention.  grid = 2048, block = 256 (8 waves).
// ---------------------------------------------------------------------------
__global__ __launch_bounds__(256, 1)
void fused_window_attn(const float* __restrict__ x,
                       const float* __restrict__ pos_emb,
                       const float* __restrict__ b_out,
                       const bf16* __restrict__ wqkvT,
                       const bf16* __restrict__ woutT,
                       float* __restrict__ out) {
  extern __shared__ char smem[];
  bf16*  xb   = (bf16*)(smem + OFF_XB);   // [64][XSTR] x (then reused as attn-out)
  bf16*  qh   = (bf16*)(smem + OFF_Q);    // [8][64][QSTR]
  bf16*  kh   = (bf16*)(smem + OFF_K);    // [8][64][QSTR]
  bf16*  vth  = (bf16*)(smem + OFF_VT);   // [8][32][VSTR]  (v transposed)
  bf16*  pb   = (bf16*)(smem + OFF_PB);   // [8 waves][16][PSTR]
  float* pos  = (float*)(smem + OFF_POS); // [15*15]

  const int tid  = threadIdx.x;
  const int wave = tid >> 5;
  const int lane = tid & 31;
  const int ln   = lane & 15;
  const int hi   = (lane >> 4) & 1;

  const int wid = blockIdx.x;             // window id 0..2047
  const int bl  = wid >> 8;               // batch*l image index
  const int win = wid & 255;
  const int wy  = win >> 4;
  const int wx  = win & 15;
  const size_t gbase = (((size_t)bl * 128 + wy * 8) * 128 + wx * 8) * DIM;

  const v8f vzero = {0.f, 0.f, 0.f, 0.f, 0.f, 0.f, 0.f, 0.f};

  // Warm L2->L0 path for this wave's first weight rows while others load x.
  __builtin_prefetch(wqkvT + (size_t)(wave * 6 * 16 + ln) * DIM, 0, 1);

  // ---- Phase 1: stage x window (64 tokens x 256 dims) into LDS as bf16 ----
  // Non-temporal loads: x is touched exactly once, keep it out of L2.
  {
    const int t  = tid >> 2;              // token 0..63
    const int dq = (tid & 3) * 64;        // 64-dim quarter
    const int ty = t >> 3, tx = t & 7;
    const v4f* src = (const v4f*)(x + gbase + ((size_t)ty * 128 + tx) * DIM + dq);
    bf16* dst = xb + t * XSTR + dq;
#pragma unroll
    for (int i = 0; i < 8; ++i) {
      v4f f0 = __builtin_nontemporal_load(src + 2 * i);
      v4f f1 = __builtin_nontemporal_load(src + 2 * i + 1);
      v8bf h;
      h[0] = (bf16)f0[0]; h[1] = (bf16)f0[1]; h[2] = (bf16)f0[2]; h[3] = (bf16)f0[3];
      h[4] = (bf16)f1[0]; h[5] = (bf16)f1[1]; h[6] = (bf16)f1[2]; h[7] = (bf16)f1[3];
      *(v8bf*)(dst + i * 8) = h;
    }
    if (tid < 225) pos[tid] = pos_emb[tid];
  }
  __syncthreads();

  // ---- Phase 2: QKV GEMM  (64 x 256) @ (256 x 768), scatter to q/k/vT ----
  // Each wave owns 6 N-tiles; B streamed from L2, A from LDS.
  const float qscale = 0.17677669529663687f;   // dh^-0.5
#pragma unroll
  for (int ntl = 0; ntl < 6; ++ntl) {
    const int nt   = wave * 6 + ntl;           // 0..47
    const int ncol = nt * 16 + ln;
    v8f acc[4] = {vzero, vzero, vzero, vzero};
    const bf16* bp = wqkvT + (size_t)ncol * DIM;
    for (int k = 0; k < 8; ++k) {
      const bf16* b0 = bp + k * 32 + hi * 8;
      v16bf bfrag = ld_frag2(b0, b0 + 16);
#pragma unroll
      for (int m = 0; m < 4; ++m) {
        const bf16* a0 = xb + (m * 16 + ln) * XSTR + k * 32 + hi * 8;
        acc[m] = wmma_bf16(ld_frag2(a0, a0 + 16), bfrag, acc[m]);
      }
    }
    const int part = nt / 16;                  // 0=q 1=k 2=v (uniform per tile)
    const int cin  = (nt & 15) * 16;           // column base within 256
    const int head = cin >> 5;                 // uniform per tile
    const int db   = (cin & 31) + ln;          // dh index for this lane
#pragma unroll
    for (int m = 0; m < 4; ++m) {
#pragma unroll
      for (int r = 0; r < 8; ++r) {
        const int tok = m * 16 + r + hi * 8;
        const float v = acc[m][r];
        if (part == 0)      qh[(head * WSQ + tok) * QSTR + db] = (bf16)(v * qscale);
        else if (part == 1) kh[(head * WSQ + tok) * QSTR + db] = (bf16)v;
        else                vth[(head * DH + db) * VSTR + tok] = (bf16)v;
      }
    }
  }
  __syncthreads();

  // ---- Phase 3: attention; wave w handles head w ----
  {
    const int head = wave;
    const bf16* qb = qh  + (size_t)head * WSQ * QSTR;
    const bf16* kb = kh  + (size_t)head * WSQ * QSTR;
    const bf16* vb = vth + (size_t)head * DH  * VSTR;
    bf16*       pw = pb  + (size_t)wave * 16  * PSTR;

#pragma unroll
    for (int mt = 0; mt < 4; ++mt) {
      // scores strip: 16 queries x 64 keys (K = dh = 32 -> one WMMA per tile)
      v8f dots[4];
      {
        const bf16* a0 = qb + (mt * 16 + ln) * QSTR + hi * 8;
        v16bf afrag = ld_frag2(a0, a0 + 16);
#pragma unroll
        for (int nt2 = 0; nt2 < 4; ++nt2) {
          const bf16* b0 = kb + (nt2 * 16 + ln) * QSTR + hi * 8;
          dots[nt2] = wmma_bf16(afrag, ld_frag2(b0, b0 + 16), vzero);
        }
      }
      // relative-position bias + row softmax (rows split across 16-lane halves)
#pragma unroll
      for (int r = 0; r < 8; ++r) {
        const int qrow = mt * 16 + r + hi * 8;
        const int qy = qrow >> 3, qx = qrow & 7;
        float vals[4];
#pragma unroll
        for (int nt2 = 0; nt2 < 4; ++nt2) {
          const int krow = nt2 * 16 + ln;
          const int ky = krow >> 3, kx = krow & 7;
          vals[nt2] = dots[nt2][r] + pos[(qy - ky + 7) * 15 + (qx - kx + 7)];
        }
        float mx = fmaxf(fmaxf(vals[0], vals[1]), fmaxf(vals[2], vals[3]));
        mx = fmaxf(mx, __shfl_xor(mx, 1));
        mx = fmaxf(mx, __shfl_xor(mx, 2));
        mx = fmaxf(mx, __shfl_xor(mx, 4));
        mx = fmaxf(mx, __shfl_xor(mx, 8));
        float s = 0.f;
#pragma unroll
        for (int nt2 = 0; nt2 < 4; ++nt2) {
          vals[nt2] = __expf(vals[nt2] - mx);
          s += vals[nt2];
        }
        s += __shfl_xor(s, 1); s += __shfl_xor(s, 2);
        s += __shfl_xor(s, 4); s += __shfl_xor(s, 8);
        const float rinv = __builtin_amdgcn_rcpf(s);
        const int prow = r + hi * 8;
#pragma unroll
        for (int nt2 = 0; nt2 < 4; ++nt2)
          pw[prow * PSTR + nt2 * 16 + ln] = (bf16)(vals[nt2] * rinv);
      }
      // LDS writes -> reads within same wave; block compiler reordering
      asm volatile("s_wait_dscnt 0" ::: "memory");

      // attn @ V : (16 x 64) @ (64 x 32), K split into two 32-steps
      v8f oacc[2] = {vzero, vzero};
#pragma unroll
      for (int kt = 0; kt < 2; ++kt) {
        const bf16* a0 = pw + ln * PSTR + kt * 32 + hi * 8;
        v16bf afrag = ld_frag2(a0, a0 + 16);
#pragma unroll
        for (int nv = 0; nv < 2; ++nv) {
          const bf16* b0 = vb + (nv * 16 + ln) * VSTR + kt * 32 + hi * 8;
          oacc[nv] = wmma_bf16(afrag, ld_frag2(b0, b0 + 16), oacc[nv]);
        }
      }
      // store head output into attn-out buffer (reuses xb)
#pragma unroll
      for (int nv = 0; nv < 2; ++nv) {
#pragma unroll
        for (int r = 0; r < 8; ++r) {
          const int tok = mt * 16 + r + hi * 8;
          xb[tok * XSTR + head * DH + nv * 16 + ln] = (bf16)oacc[nv][r];
        }
      }
    }
  }
  __syncthreads();

  // ---- Phase 4: output projection (64 x 256) @ (256 x 256) + bias ----
#pragma unroll
  for (int ntl = 0; ntl < 2; ++ntl) {
    const int nt   = wave * 2 + ntl;            // 0..15
    const int ncol = nt * 16 + ln;
    v8f acc[4] = {vzero, vzero, vzero, vzero};
    const bf16* bp = woutT + (size_t)ncol * DIM;
    for (int k = 0; k < 8; ++k) {
      const bf16* b0 = bp + k * 32 + hi * 8;
      v16bf bfrag = ld_frag2(b0, b0 + 16);
#pragma unroll
      for (int m = 0; m < 4; ++m) {
        const bf16* a0 = xb + (m * 16 + ln) * XSTR + k * 32 + hi * 8;
        acc[m] = wmma_bf16(ld_frag2(a0, a0 + 16), bfrag, acc[m]);
      }
    }
    const float bo = b_out[ncol];
    // Non-temporal stores: out is written once and never re-read on device.
#pragma unroll
    for (int m = 0; m < 4; ++m) {
#pragma unroll
      for (int r = 0; r < 8; ++r) {
        const int tok = m * 16 + r + hi * 8;
        const int ty = tok >> 3, tx = tok & 7;
        __builtin_nontemporal_store(
            acc[m][r] + bo,
            out + gbase + ((size_t)ty * 128 + tx) * DIM + ncol);
      }
    }
  }
}

// ---------------------------------------------------------------------------
extern "C" void kernel_launch(void* const* d_in, const int* in_sizes, int n_in,
                              void* d_out, int out_size, void* d_ws, size_t ws_size,
                              hipStream_t stream) {
  const float* x       = (const float*)d_in[0];
  const float* w_qkv   = (const float*)d_in[1];
  const float* pos_emb = (const float*)d_in[2];
  const float* w_out   = (const float*)d_in[3];
  const float* b_out   = (const float*)d_in[4];
  float*       out     = (float*)d_out;

  bf16* wqkvT = (bf16*)d_ws;                       // 768*256 bf16 = 384 KB
  bf16* woutT = wqkvT + (size_t)NQKV * DIM;        // 256*256 bf16 = 128 KB

  prep_weights<<<NQKV + DIM, 256, 0, stream>>>(w_qkv, w_out, wqkvT, woutT);

  fused_window_attn<<<2048, 256, SMEM_BYTES, stream>>>(
      x, pos_emb, b_out, wqkvT, woutT, out);
}
